// SelfDotProductAttnQKNorm_27977416966338
// MI455X (gfx1250) — matrix operand order repaired and checked by reading, not compiled, not run on hardware
//
#include <hip/hip_runtime.h>

typedef __attribute__((ext_vector_type(16))) _Float16 v16h;
typedef __attribute__((ext_vector_type(8)))  _Float16 v8h;
typedef __attribute__((ext_vector_type(2)))  __fp16   v2hp;  // type returned by cvt_pkrtz
typedef __attribute__((ext_vector_type(8)))  float    v8f;

union AFrag { v16h v; v8h h[2]; };

static constexpr int Bb = 2, Hh = 16, Ss = 2048, Dd = 64;
static constexpr int BQ = 128;   // query rows per block (8 waves x 16)
static constexpr int BKV = 32;   // keys per iteration

// v_mov_b32_dpp cross-lane move (VALU, no LDS, no waits)
template<int CTRL>
__device__ __forceinline__ float dppmov(float x) {
  return __int_as_float(
      __builtin_amdgcn_update_dpp(0, __float_as_int(x), CTRL, 0xF, 0xF, true));
}
static constexpr int DPP_XOR1 = 0xB1;   // quad_perm(1,0,3,2)
static constexpr int DPP_XOR2 = 0x4E;   // quad_perm(2,3,0,1)
static constexpr int DPP_HMIR = 0x141;  // row_half_mirror (8-lane swap)
static constexpr int DPP_MIR  = 0x140;  // row_mirror (16-lane swap)

__launch_bounds__(256)
__global__ void fa_qknorm_kernel(const float* __restrict__ Q,
                                 const float* __restrict__ K,
                                 const float* __restrict__ V,
                                 const float* __restrict__ gainp,
                                 const int*   __restrict__ causalp,
                                 float* __restrict__ Out)
{
  __shared__ _Float16 sQ[BQ * Dd];         // 16 KB: normalized Q, row-major
  __shared__ _Float16 sK[BKV * Dd];        //  4 KB: normalized K, row-major [key][d]
  __shared__ _Float16 sVt[Dd * BKV];       //  4 KB: V transposed [d][key-slot]
  __shared__ _Float16 sP[8 * 16 * BKV];    //  8 KB: per-wave P tiles [row][key-slot]

  const int tid  = threadIdx.x;
  const int wave = tid >> 5;
  const int lane = tid & 31;
  const int l16  = lane & 15;
  const int hi   = lane >> 4;

  const int qb = blockIdx.x * BQ;
  const int bh = blockIdx.y;

  const float* Qh = Q + (size_t)bh * Ss * Dd;
  const float* Kh = K + (size_t)bh * Ss * Dd;
  const float* Vh = V + (size_t)bh * Ss * Dd;

  const float gain   = gainp[0];
  const int   causal = causalp[0];
  const float Cs = gain * 1.4426950408889634f;  // gain * log2(e)

  // ---- stage L2-normalized Q (f32 -> f16) into LDS: 2 threads per row ----
  {
    const int row  = tid >> 1;
    const int half = tid & 1;
    const float4* qp = (const float4*)(Qh + (size_t)(qb + row) * Dd + half * 32);
    float4 q[8];
    float ss = 0.f;
#pragma unroll
    for (int j = 0; j < 8; ++j) {
      q[j] = qp[j];
      ss += q[j].x*q[j].x + q[j].y*q[j].y + q[j].z*q[j].z + q[j].w*q[j].w;
    }
    ss += dppmov<DPP_XOR1>(ss);           // pair reduction, lanes 2k<->2k+1
    const float rn = rsqrtf(ss + 1e-12f);
    _Float16* dq = sQ + row * Dd + half * 32;
#pragma unroll
    for (int j = 0; j < 8; ++j) {
      dq[j*4+0] = (_Float16)(q[j].x * rn);
      dq[j*4+1] = (_Float16)(q[j].y * rn);
      dq[j*4+2] = (_Float16)(q[j].z * rn);
      dq[j*4+3] = (_Float16)(q[j].w * rn);
    }
  }
  __syncthreads();

  // ---- per-wave Q A-fragments: 16 rows x 64d = two 16x32 f16 frags ----
  AFrag qf[2];
  {
    const int qbase = (wave*16 + l16) * Dd + hi*8;
#pragma unroll
    for (int c = 0; c < 2; ++c) {
      qf[c].h[0] = *(const v8h*)(sQ + qbase + c*32);
      qf[c].h[1] = *(const v8h*)(sQ + qbase + c*32 + 16);
    }
  }

  v8f acc[4] = {};        // 16 rows x 64 cols f32
  float lpart[8];         // per-lane partial row sums (cols owned by this lane)
#pragma unroll
  for (int v = 0; v < 8; ++v) lpart[v] = 0.f;

  const int qrow0   = qb + wave*16 + 8*hi;
  // wave-uniform first row, forced into an SGPR so tile classification becomes
  // scalar branches (s_cmp/s_cbranch) and WMMA provably runs with EXEC all-1s
  const int waveRow = __builtin_amdgcn_readfirstlane(qb + wave*16);
  const int kv_end  = causal ? (qb + BQ) : Ss;

  for (int kv = 0; kv < kv_end; kv += BKV) {
    __syncthreads();             // prior iteration's LDS reads complete

    // ---- stage normalized K (row-major) + V (transposed, key-interleaved) ----
    {
      const int row = tid >> 3;                    // key within tile
      const int j   = tid & 7;                     // 8 d-elems per thread
      const int slot = ((row & 15) << 1) | (row >> 4);   // interleaved key slot
      const float* kp = Kh + (size_t)(kv + row) * Dd + j*8;
      float4 k0 = *(const float4*)(kp);
      float4 k1 = *(const float4*)(kp + 4);
      float ss = k0.x*k0.x + k0.y*k0.y + k0.z*k0.z + k0.w*k0.w
               + k1.x*k1.x + k1.y*k1.y + k1.z*k1.z + k1.w*k1.w;
      ss += dppmov<DPP_XOR1>(ss);
      ss += dppmov<DPP_XOR2>(ss);
      ss += dppmov<DPP_HMIR>(ss);                  // full 8-lane (one-row) sum
      const float rn = rsqrtf(ss + 1e-12f);
      _Float16* dk = sK + row * Dd + j*8;
      dk[0] = (_Float16)(k0.x*rn); dk[1] = (_Float16)(k0.y*rn);
      dk[2] = (_Float16)(k0.z*rn); dk[3] = (_Float16)(k0.w*rn);
      dk[4] = (_Float16)(k1.x*rn); dk[5] = (_Float16)(k1.y*rn);
      dk[6] = (_Float16)(k1.z*rn); dk[7] = (_Float16)(k1.w*rn);

      const float* vp = Vh + (size_t)(kv + row) * Dd + j*8;
      float4 w0 = *(const float4*)(vp);
      float4 w1 = *(const float4*)(vp + 4);
      sVt[(j*8+0)*BKV + slot] = (_Float16)w0.x;
      sVt[(j*8+1)*BKV + slot] = (_Float16)w0.y;
      sVt[(j*8+2)*BKV + slot] = (_Float16)w0.z;
      sVt[(j*8+3)*BKV + slot] = (_Float16)w0.w;
      sVt[(j*8+4)*BKV + slot] = (_Float16)w1.x;
      sVt[(j*8+5)*BKV + slot] = (_Float16)w1.y;
      sVt[(j*8+6)*BKV + slot] = (_Float16)w1.z;
      sVt[(j*8+7)*BKV + slot] = (_Float16)w1.w;

      if (kv + BKV < kv_end) {   // prefetch next tile (global_prefetch_b8)
        __builtin_prefetch(Kh + (size_t)(kv + BKV + row) * Dd + j*8, 0, 1);
        __builtin_prefetch(Vh + (size_t)(kv + BKV + row) * Dd + j*8, 0, 1);
      }
    }
    __syncthreads();

    // scalar (SGPR) tile classification
    const bool fullMask = causal && (kv > waveRow + 15);          // P == 0, skip all
    const bool diagTile = causal && (kv + BKV - 1 > waveRow);     // needs elementwise mask

    if (!fullMask) {
      // ---- load ALL K fragments first (8x ds_load_b128, overlapped) ----
      AFrag kf[2][2];
#pragma unroll
      for (int t = 0; t < 2; ++t)
#pragma unroll
        for (int ch = 0; ch < 2; ++ch) {
          const int kb = (t*16 + l16) * Dd + ch*32 + hi*8;
          kf[t][ch].h[0] = *(const v8h*)(sK + kb);
          kf[t][ch].h[1] = *(const v8h*)(sK + kb + 16);
        }

      // ---- scores: S = Qn * Kn^T, two 16x16 f32 tiles, K-dim = 64 ----
      v8f sc[2];
#pragma unroll
      for (int t = 0; t < 2; ++t) {
        v8f c = {};
        c = __builtin_amdgcn_wmma_f32_16x16x32_f16(false, qf[0].v, false, kf[t][0].v,
                                                   (short)0, c, false, false);
        c = __builtin_amdgcn_wmma_f32_16x16x32_f16(false, qf[1].v, false, kf[t][1].v,
                                                   (short)0, c, false, false);
        sc[t] = c;
      }

      // ---- fixed-shift softmax (cos-sim scores bounded by 1 => shift = 1) ----
      _Float16* pw = sP + wave * (16*BKV);
      if (diagTile) {
#pragma unroll
        for (int v = 0; v < 8; ++v) {
          float a0 = sc[0][v], a1 = sc[1][v];
          const int rowg = qrow0 + v;
          if (kv + l16 > rowg)      a0 = -1e30f;   // v_exp_f32(-huge) == 0
          if (kv + 16 + l16 > rowg) a1 = -1e30f;
          const float p0 = __builtin_amdgcn_exp2f(__builtin_fmaf(Cs, a0, -Cs));
          const float p1 = __builtin_amdgcn_exp2f(__builtin_fmaf(Cs, a1, -Cs));
          lpart[v] += p0 + p1;
          const v2hp pk = __builtin_amdgcn_cvt_pkrtz(p0, p1);
          *(v2hp*)(pw + (v + 8*hi)*BKV + l16*2) = pk;       // one ds_store_b32
        }
      } else {
#pragma unroll
        for (int v = 0; v < 8; ++v) {
          const float p0 = __builtin_amdgcn_exp2f(__builtin_fmaf(Cs, sc[0][v], -Cs));
          const float p1 = __builtin_amdgcn_exp2f(__builtin_fmaf(Cs, sc[1][v], -Cs));
          lpart[v] += p0 + p1;
          const v2hp pk = __builtin_amdgcn_cvt_pkrtz(p0, p1);
          *(v2hp*)(pw + (v + 8*hi)*BKV + l16*2) = pk;
        }
      }
    }
    __syncthreads();             // P visible; sK reads done (uniform barrier)

    if (!fullMask) {
      // ---- load P + ALL V fragments first, then 4 WMMAs ----
      AFrag pf;
      const _Float16* pw = sP + wave * (16*BKV);
      const int pb = l16*BKV + hi*8;
      pf.h[0] = *(const v8h*)(pw + pb);
      pf.h[1] = *(const v8h*)(pw + pb + 16);
      AFrag vfr[4];
#pragma unroll
      for (int n = 0; n < 4; ++n) {
        const int vb = (n*16 + l16)*BKV + hi*8;
        vfr[n].h[0] = *(const v8h*)(sVt + vb);
        vfr[n].h[1] = *(const v8h*)(sVt + vb + 16);
      }
#pragma unroll
      for (int n = 0; n < 4; ++n)
        acc[n] = __builtin_amdgcn_wmma_f32_16x16x32_f16(false, pf.v, false, vfr[n].v,
                                                        (short)0, acc[n], false, false);
    }
  }

  // ---- one-time row-sum reduction across 16 lanes, normalize, store ----
  float* Oh = Out + (size_t)bh * Ss * Dd;
#pragma unroll
  for (int v = 0; v < 8; ++v) {
    float l = lpart[v];
    l += dppmov<DPP_XOR1>(l);
    l += dppmov<DPP_XOR2>(l);
    l += dppmov<DPP_HMIR>(l);
    l += dppmov<DPP_MIR>(l);     // full 16-lane row sum
    const float inv = 1.0f / l;
    float* orow = Oh + (size_t)(qrow0 + v) * Dd;
#pragma unroll
    for (int n = 0; n < 4; ++n)
      orow[n*16 + l16] = acc[n][v] * inv;
  }
}

extern "C" void kernel_launch(void* const* d_in, const int* in_sizes, int n_in,
                              void* d_out, int out_size, void* d_ws, size_t ws_size,
                              hipStream_t stream) {
  (void)in_sizes; (void)n_in; (void)out_size; (void)d_ws; (void)ws_size;
  const float* Q = (const float*)d_in[0];
  const float* K = (const float*)d_in[1];
  const float* V = (const float*)d_in[2];
  const float* g = (const float*)d_in[3];
  const int*   c = (const int*)d_in[4];
  float* out = (float*)d_out;
  dim3 grid(Ss / BQ, Bb * Hh);
  fa_qknorm_kernel<<<grid, dim3(256), 0, stream>>>(Q, K, V, g, c, out);
}
